// SimpleRNN_7834020348317
// MI455X (gfx1250) — matrix operand order repaired and checked by reading, not compiled, not run on hardware
//
#include <hip/hip_runtime.h>
#include <hip/hip_bf16.h>

// Problem dims (match the JAX reference)
#define B_   64
#define T_   512
#define DIN  256
#define DH   1024
#define DOUT 256

typedef __bf16 bf16;
typedef __attribute__((ext_vector_type(16))) __bf16 v16bf;
typedef __attribute__((ext_vector_type(8)))  float  v8f;

// ---------------------------------------------------------------------------
// WMMA A fragment loader, per CDNA5 ISA 7.12.2 VGPR layout (wave32).
//   lanes 0-15  : M = lane,    halves 0..7 = K 0..7,  halves 8..15 = K 16..23
//   lanes 16-31 : M = lane-16, halves 0..7 = K 8..15, halves 8..15 = K 24..31
// On LDS-resident tiles clang emits ds_load_b128 x2.
__device__ __forceinline__ v16bf load_a16x32(const bf16* __restrict__ A, int lda) {
    const int lane = threadIdx.x & 31;
    const int row  = lane & 15;
    const int kh   = (lane >> 4) << 3;          // 0 or 8
    const bf16* p  = A + (long)row * lda;
    v16bf a;
#pragma unroll
    for (int e = 0; e < 8; ++e) {
        a[e]     = p[kh + e];
        a[e + 8] = p[16 + kh + e];
    }
    return a;
}

// B fragments are pre-packed so each lane's 16 halves are contiguous (32B):
//   lanes 0-15  : N = lane,    e 0..7 = K 0..7,  e 8..15 = K 16..23 of the 32-wide k-tile
//   lanes 16-31 : N = lane-16, e 0..7 = K 8..15, e 8..15 = K 24..31
__device__ __forceinline__ v16bf load_b_packed(const bf16* __restrict__ P) {
    const int lane = threadIdx.x & 31;
    return ((const v16bf*)P)[lane];
}

// ---------------------------------------------------------------------------
// Async staging of a contiguous global panel into LDS (CDNA5 async path):
// GLOBAL_LOAD_ASYNC_TO_LDS_B128 (ASYNCcnt), s_wait_asynccnt 0, block barrier.
// Generic LDS pointer low 32 bits == LDS byte address (ISA 10.2 aperture rule).
// DEVSCOPE=true forces SCOPE_DEV so staging is coherent with other WGPs'
// stores from the previous scan step.
template <bool DEVSCOPE>
__device__ __forceinline__ void stage_async(const bf16* __restrict__ gsrc,
                                            bf16* lds_dst, int nbytes) {
    const unsigned lbase = (unsigned)(unsigned long long)lds_dst;
    for (int c = threadIdx.x * 16; c < nbytes; c += 128 * 16) {
        unsigned la           = lbase + c;
        unsigned long long ga = (unsigned long long)gsrc + c;
        if (DEVSCOPE)
            asm volatile("global_load_async_to_lds_b128 %0, %1, off scope:SCOPE_DEV"
                         :: "v"(la), "v"(ga) : "memory");
        else
            asm volatile("global_load_async_to_lds_b128 %0, %1, off"
                         :: "v"(la), "v"(ga) : "memory");
    }
    asm volatile("s_wait_asynccnt 0x0" ::: "memory");
    __syncthreads();
}

// ---------------------------------------------------------------------------
// f32 -> bf16 bulk convert (input activations)
__global__ void __launch_bounds__(256)
cvt_f32_to_bf16(const float* __restrict__ src, bf16* __restrict__ dst, int n) {
    for (int i = blockIdx.x * blockDim.x + threadIdx.x; i < n;
         i += gridDim.x * blockDim.x)
        dst[i] = (bf16)src[i];
}

// ---------------------------------------------------------------------------
// Pack a row-major f32 weight [K x N] into bf16 WMMA-B fragment order.
__global__ void __launch_bounds__(256)
pack_b_frag(const float* __restrict__ W, bf16* __restrict__ P, int K, int N) {
    const int p = blockIdx.x * blockDim.x + threadIdx.x;
    if (p >= K * N) return;
    const int e      = p & 15;
    const int lane   = (p >> 4) & 31;
    const int tile   = p >> 9;                  // / 512 halves per fragment
    const int ktiles = K >> 5;
    const int tk     = tile % ktiles;
    const int tn     = tile / ktiles;
    const int col    = lane & 15;
    const int kk     = tk * 32 + ((lane >> 4) << 3) + (e & 7) + ((e >> 3) << 4);
    P[p] = (bf16)W[(long)kk * N + tn * 16 + col];
}

// ---------------------------------------------------------------------------
// Zero the grid-barrier state (makes kernel_launch deterministic per call).
__global__ void init_sync(unsigned* __restrict__ sync) {
    if (threadIdx.x < 2) sync[threadIdx.x] = 0u;
}

// ---------------------------------------------------------------------------
// Phase 1: x_proj[t,b,h] = X[b,t,:] @ Wi[:,h] + hidden_bias[h]  (f32, time-major)
// grid = (DH/64, B*T/16), block = 128 (4 waves, one 16x16 tile each, shared A)
__global__ void __launch_bounds__(128)
xproj_gemm(const bf16* __restrict__ X, const bf16* __restrict__ WiP,
           const float* __restrict__ hbias, float* __restrict__ xproj) {
    __shared__ __align__(16) bf16 tileA[16 * DIN];        // 8 KB
    const int wave = threadIdx.x >> 5;
    const int lane = threadIdx.x & 31;
    const int tn   = blockIdx.x * 4 + wave;               // 0..63
    const int tm   = blockIdx.y;                          // 0..2047 (flat b*T+t)

    stage_async<false>(X + (long)tm * 16 * DIN, tileA, 16 * DIN * (int)sizeof(bf16));

    const int col = tn * 16 + (lane & 15);
    v8f acc;
    const float bias = hbias[col];
#pragma unroll
    for (int v = 0; v < 8; ++v) acc[v] = bias;

    const bf16* Bbase = WiP + ((long)tn * (DIN / 32) << 9);
#pragma unroll
    for (int tk = 0; tk < DIN / 32; ++tk) {
        __builtin_prefetch(Bbase + ((tk + 1) << 9), 0, 0);
        v16bf a = load_a16x32(tileA + tk * 32, DIN);
        v16bf b = load_b_packed(Bbase + (tk << 9));
        acc = __builtin_amdgcn_wmma_f32_16x16x32_bf16(
                  false, a, false, b, (short)0, acc, false, false);
    }

    const int rbase = (lane >> 4) * 8;
#pragma unroll
    for (int v = 0; v < 8; ++v) {
        const int bt = tm * 16 + rbase + v;
        const int bb = bt >> 9;                 // / T_
        const int tt = bt & (T_ - 1);
        xproj[((long)tt * B_ + bb) * DH + col] = acc[v];
    }
}

// ---------------------------------------------------------------------------
// Phase 2 (critical path): ONE persistent kernel for the whole scan.
//   h_t = tanh(x_proj[t] + h_{t-1} @ Wh), hidden bf16 time-major [T,B,H].
// grid = (16, 4) = 64 blocks — small enough to co-reside; the per-step
// device-wide barrier replaces 512 kernel launches (launch overhead would
// otherwise dominate: per-step math is ~2 us, Wh is 2 MB L2-resident).
__global__ void __launch_bounds__(128)
rnn_scan(const bf16* __restrict__ WhP, const float* __restrict__ xproj,
         bf16* __restrict__ hidden, unsigned* __restrict__ sync) {
    __shared__ __align__(16) bf16 tileA[16 * DH];         // 32 KB
    const int wave  = threadIdx.x >> 5;
    const int lane  = threadIdx.x & 31;
    const int tn    = blockIdx.x * 4 + wave;              // 0..63
    const int tm    = blockIdx.y;                         // 0..3
    const int col   = tn * 16 + (lane & 15);
    const int rbase = (lane >> 4) * 8;
    const bf16* Bbase = WhP + ((long)tn * (DH / 32) << 9);
    unsigned* cnt = sync;
    unsigned* gen = sync + 1;
    const unsigned nblocks = gridDim.x * gridDim.y;       // 64

    for (int t = 0; t < T_; ++t) {
        if (t > 0)   // stage h_{t-1}[tm*16 .. +15, :] (contiguous 32 KB, L2-hot)
            stage_async<true>(hidden + ((long)(t - 1) * B_ + tm * 16) * DH,
                              tileA, 16 * DH * (int)sizeof(bf16));

        // Accumulator seeded with x_proj (already includes hidden_bias), f32.
        v8f acc;
#pragma unroll
        for (int v = 0; v < 8; ++v) {
            const int brow = tm * 16 + rbase + v;
            acc[v] = xproj[((long)t * B_ + brow) * DH + col];
        }

        if (t > 0) {
#pragma unroll 4
            for (int tk = 0; tk < DH / 32; ++tk) {
                __builtin_prefetch(Bbase + ((tk + 1) << 9), 0, 3);
                v16bf a = load_a16x32(tileA + tk * 32, DH);   // ds_load_b128 x2
                v16bf b = load_b_packed(Bbase + (tk << 9));   // 32B coalesced
                acc = __builtin_amdgcn_wmma_f32_16x16x32_bf16(
                          false, a, false, b, (short)0, acc, false, false);
            }
        }

        // tanh in f32, requantize h_t to bf16 (time-major)
#pragma unroll
        for (int v = 0; v < 8; ++v) {
            const int brow = tm * 16 + rbase + v;
            hidden[((long)t * B_ + brow) * DH + col] = (bf16)tanhf(acc[v]);
        }

        // ---- device-wide barrier: release h_t, await all 64 blocks ----
        __threadfence();                 // each wave: wait own stores, WB to DEV
        __syncthreads();
        if (threadIdx.x == 0) {
            const unsigned target = nblocks * (unsigned)(t + 1);
            const unsigned ticket = __hip_atomic_fetch_add(
                cnt, 1u, __ATOMIC_ACQ_REL, __HIP_MEMORY_SCOPE_AGENT);
            if (ticket == target - 1u)
                __hip_atomic_store(gen, (unsigned)(t + 1),
                                   __ATOMIC_RELEASE, __HIP_MEMORY_SCOPE_AGENT);
            else
                while (__hip_atomic_load(gen, __ATOMIC_ACQUIRE,
                                         __HIP_MEMORY_SCOPE_AGENT) < (unsigned)(t + 1))
                    __builtin_amdgcn_s_sleep(2);
        }
        __syncthreads();
        __threadfence();                 // acquire: invalidate stale cache lines
    }
}

// ---------------------------------------------------------------------------
// Phase 3: out[b,t,o] = H[t,b,:] @ Wo[:,o] + output_bias[o]   (f32 out)
// grid = (DOUT/64, T*B/16), block = 128; m-tiles over time-major flat rows.
__global__ void __launch_bounds__(128)
out_gemm(const bf16* __restrict__ H, const bf16* __restrict__ WoP,
         const float* __restrict__ obias, float* __restrict__ out) {
    __shared__ __align__(16) bf16 tileA[16 * DH];         // 32 KB
    const int wave = threadIdx.x >> 5;
    const int lane = threadIdx.x & 31;
    const int tn   = blockIdx.x * 4 + wave;               // 0..15
    const int tm   = blockIdx.y;                          // 0..2047 (flat t*B+b)

    stage_async<false>(H + (long)tm * 16 * DH, tileA, 16 * DH * (int)sizeof(bf16));

    const int col = tn * 16 + (lane & 15);
    v8f acc;
    const float bias = obias[col];
#pragma unroll
    for (int v = 0; v < 8; ++v) acc[v] = bias;

    const bf16* Bbase = WoP + ((long)tn * (DH / 32) << 9);
#pragma unroll 4
    for (int tk = 0; tk < DH / 32; ++tk) {
        __builtin_prefetch(Bbase + ((tk + 1) << 9), 0, 0);
        v16bf a = load_a16x32(tileA + tk * 32, DH);
        v16bf b = load_b_packed(Bbase + (tk << 9));
        acc = __builtin_amdgcn_wmma_f32_16x16x32_bf16(
                  false, a, false, b, (short)0, acc, false, false);
    }

    // flat row tb -> (t = tb/B, b = tb%B) -> out[b][t][col]   (B_ == 64)
    const int rbase = (lane >> 4) * 8;
#pragma unroll
    for (int v = 0; v < 8; ++v) {
        const int tb = tm * 16 + rbase + v;
        const int tt = tb >> 6;
        const int bb = tb & (B_ - 1);
        out[((long)bb * T_ + tt) * DOUT + col] = acc[v];
    }
}

// ---------------------------------------------------------------------------
extern "C" void kernel_launch(void* const* d_in, const int* in_sizes, int n_in,
                              void* d_out, int out_size, void* d_ws, size_t ws_size,
                              hipStream_t stream) {
    const float* inputs        = (const float*)d_in[0];  // [B,T,DIN]
    const float* input_kernel  = (const float*)d_in[1];  // [DIN,DH]
    const float* hidden_kernel = (const float*)d_in[2];  // [DH,DH]
    const float* hidden_bias   = (const float*)d_in[3];  // [DH]
    const float* output_kernel = (const float*)d_in[4];  // [DH,DOUT]
    const float* output_bias   = (const float*)d_in[5];  // [DOUT]
    float* out = (float*)d_out;                          // [B,T,DOUT]

    // Workspace carve-up (~211 MB total)
    char* w = (char*)d_ws;
    float* xproj  = (float*)w; w += (size_t)T_ * B_ * DH  * sizeof(float); // 128 MB, [T,B,H]
    bf16*  hidden = (bf16*) w; w += (size_t)T_ * B_ * DH  * sizeof(bf16);  //  64 MB, [T,B,H]
    bf16*  xbf    = (bf16*) w; w += (size_t)B_ * T_ * DIN * sizeof(bf16);  //  16 MB
    bf16*  wiP    = (bf16*) w; w += (size_t)DIN * DH      * sizeof(bf16);
    bf16*  whP    = (bf16*) w; w += (size_t)DH  * DH      * sizeof(bf16);
    bf16*  woP    = (bf16*) w; w += (size_t)DH  * DOUT    * sizeof(bf16);
    unsigned* sync = (unsigned*)w;                       // 2 words, barrier state

    // Reset grid-barrier state; quantize activations; pack weights
    init_sync<<<1, 32, 0, stream>>>(sync);
    {
        const int n0 = B_ * T_ * DIN;
        cvt_f32_to_bf16<<<(n0 + 255) / 256, 256, 0, stream>>>(inputs, xbf, n0);
        const int n1 = DIN * DH;
        pack_b_frag<<<(n1 + 255) / 256, 256, 0, stream>>>(input_kernel, wiP, DIN, DH);
        const int n2 = DH * DH;
        pack_b_frag<<<(n2 + 255) / 256, 256, 0, stream>>>(hidden_kernel, whP, DH, DH);
        const int n3 = DH * DOUT;
        pack_b_frag<<<(n3 + 255) / 256, 256, 0, stream>>>(output_kernel, woP, DH, DOUT);
    }

    // Phase 1: input projection (+ hidden_bias) for all timesteps
    xproj_gemm<<<dim3(DH / 64, (B_ * T_) / 16), 128, 0, stream>>>(
        xbf, wiP, hidden_bias, xproj);

    // Phase 2: entire 512-step scan in ONE persistent kernel (64 blocks)
    rnn_scan<<<dim3(DH / 64, B_ / 16), 128, 0, stream>>>(whP, xproj, hidden, sync);

    // Phase 3: readout over every timestep
    out_gemm<<<dim3(DOUT / 64, (T_ * B_) / 16), 128, 0, stream>>>(
        hidden, woP, output_bias, out);
}
